// _RQVAEModel_18674517803205
// MI455X (gfx1250) — matrix operand order repaired and checked by reading, not compiled, not run on hardware
//
#include <hip/hip_runtime.h>
#include <hip/hip_fp16.h>
#include <math.h>

#define BSZ   32768
#define DDIM  768
#define HDIM  1024
#define LDIM  256
#define KCODES 2048
#define RQL 8
#define RQ_WAVES 4       // waves per block in rq kernel
#define RQ_ROWS_PER_WAVE 32
#define RQ_ROWS_PER_BLOCK (RQ_WAVES * RQ_ROWS_PER_WAVE)   /* 128 */

typedef __attribute__((ext_vector_type(16))) _Float16 v16h;
typedef __attribute__((ext_vector_type(8)))  _Float16 v8h;
typedef __attribute__((ext_vector_type(4)))  _Float16 v4h;
typedef __attribute__((ext_vector_type(8)))  float    v8f;
typedef __attribute__((ext_vector_type(4)))  float    v4f;

__device__ inline v8f zero_v8f() {
  v8f z = {0.f, 0.f, 0.f, 0.f, 0.f, 0.f, 0.f, 0.f};
  return z;
}

__device__ inline float gelu_exact(float v) {
  return 0.5f * v * (1.0f + erff(v * 0.7071067811865475f));
}

// A fragment (16x32 f16) from LDS tile, rows stored [row][32] f16.
// Lanes 0-15: row=lane, K = 0..7 (v0..3) and 16..23 (v4..7)
// Lanes 16-31: row=lane-16, K = 8..15 and 24..31
__device__ inline v16h frag_a_lds(const _Float16* tile, int row_base) {
  int lane = threadIdx.x & 31;
  const _Float16* p = tile + (row_base + (lane & 15)) * 32;
  int ks = (lane & 16) ? 8 : 0;
  v8h lo = *(const v8h*)(p + ks);
  v8h hi = *(const v8h*)(p + ks + 16);
  v16h f;
#pragma unroll
  for (int i = 0; i < 8; ++i) { f[i] = lo[i]; f[8 + i] = hi[i]; }
  return f;
}

// B fragment (32x16 f16); B stored transposed as [col][32] f16.
// Lane holds col = lane&15; lanes 0-15: K=0..7 and 16..23, lanes 16-31: K=8..15 and 24..31.
__device__ inline v16h frag_b_lds(const _Float16* tile, int col_base) {
  int lane = threadIdx.x & 31;
  const _Float16* p = tile + (col_base + (lane & 15)) * 32 + ((lane & 16) ? 8 : 0);
  v8h lo = *(const v8h*)p;
  v8h hi = *(const v8h*)(p + 16);
  v16h f;
#pragma unroll
  for (int i = 0; i < 8; ++i) { f[i] = lo[i]; f[8 + i] = hi[i]; }
  return f;
}

// ---------------------------------------------------------------------------
// Generic WMMA GEMM: Out = act(A[M,Kt] @ Wt[N,Kt]^T + bias)
// AF16: A is f16 (else f32, converted while staging)
// ACT : exact GELU
// EPI : 0 = store f16, 1 = store f32, 2 = fused sum((out - Xref)^2) -> loss_acc
// ---------------------------------------------------------------------------
template<bool AF16, bool ACT, int EPI>
__global__ __launch_bounds__(256) void gemm_kernel(
    const void* __restrict__ Ap, const _Float16* __restrict__ Wt,
    const float* __restrict__ bias, void* __restrict__ Outp,
    const float* __restrict__ Xref, float* __restrict__ loss_acc,
    int M, int N, int Kt)
{
  __shared__ _Float16 As[128 * 32];
  __shared__ _Float16 Bs[128 * 32];
  __shared__ float lred[256];

  const int tid = threadIdx.x;
  const int lane = tid & 31;
  const int wid = tid >> 5;
  const int waveM = wid & 1;   // 2 row groups of 64
  const int waveN = wid >> 1;  // 4 col groups of 32
  const size_t rowT = (size_t)blockIdx.y * 128;
  const size_t colT = (size_t)blockIdx.x * 128;

  v8f acc[4][2];
#pragma unroll
  for (int a = 0; a < 4; ++a)
#pragma unroll
    for (int b = 0; b < 2; ++b) acc[a][b] = zero_v8f();

  const int ldr = tid >> 1;        // 0..127
  const int ldk = (tid & 1) * 16;  // 0 or 16

  for (int kt = 0; kt < Kt; kt += 32) {
    if constexpr (AF16) {
      const _Float16* src = (const _Float16*)Ap + (rowT + ldr) * (size_t)Kt + kt + ldk;
      *(v8h*)(As + ldr * 32 + ldk)     = *(const v8h*)src;
      *(v8h*)(As + ldr * 32 + ldk + 8) = *(const v8h*)(src + 8);
    } else {
      const float* src = (const float*)Ap + (rowT + ldr) * (size_t)Kt + kt + ldk;
      v4f a0 = *(const v4f*)(src);
      v4f a1 = *(const v4f*)(src + 4);
      v4f a2 = *(const v4f*)(src + 8);
      v4f a3 = *(const v4f*)(src + 12);
      _Float16* d = As + ldr * 32 + ldk;
#pragma unroll
      for (int i = 0; i < 4; ++i) {
        d[i]      = (_Float16)a0[i];
        d[4 + i]  = (_Float16)a1[i];
        d[8 + i]  = (_Float16)a2[i];
        d[12 + i] = (_Float16)a3[i];
      }
    }
    {
      const _Float16* src = Wt + (colT + ldr) * (size_t)Kt + kt + ldk;
      *(v8h*)(Bs + ldr * 32 + ldk)     = *(const v8h*)src;
      *(v8h*)(Bs + ldr * 32 + ldk + 8) = *(const v8h*)(src + 8);
    }
    __syncthreads();

    v16h af[4], bf[2];
#pragma unroll
    for (int a = 0; a < 4; ++a) af[a] = frag_a_lds(As, waveM * 64 + a * 16);
#pragma unroll
    for (int b = 0; b < 2; ++b) bf[b] = frag_b_lds(Bs, waveN * 32 + b * 16);
#pragma unroll
    for (int a = 0; a < 4; ++a)
#pragma unroll
      for (int b = 0; b < 2; ++b)
        acc[a][b] = __builtin_amdgcn_wmma_f32_16x16x32_f16(
            false, af[a], false, bf[b], (short)0, acc[a][b], false, false);
    __syncthreads();
  }

  float lsum = 0.0f;
#pragma unroll
  for (int a = 0; a < 4; ++a) {
#pragma unroll
    for (int b = 0; b < 2; ++b) {
      const size_t col  = colT + waveN * 32 + b * 16 + (lane & 15);
      const size_t row0 = rowT + waveM * 64 + a * 16 + ((lane & 16) ? 8 : 0);
      const float bv = bias[col];
#pragma unroll
      for (int j = 0; j < 8; ++j) {
        float v = acc[a][b][j] + bv;
        if constexpr (ACT) v = gelu_exact(v);
        const size_t row = row0 + j;
        if constexpr (EPI == 0) {
          ((_Float16*)Outp)[row * (size_t)N + col] = (_Float16)v;
        } else if constexpr (EPI == 1) {
          ((float*)Outp)[row * (size_t)N + col] = v;
        } else {
          float dlt = v - Xref[row * (size_t)N + col];
          lsum += dlt * dlt;
        }
      }
    }
  }
  if constexpr (EPI == 2) {
    lred[tid] = lsum;
    __syncthreads();
    if (tid == 0) {
      float s = 0.0f;
      for (int i = 0; i < 256; ++i) s += lred[i];
      atomicAdd(loss_acc, s);
    }
  }
}

// ---------------------------------------------------------------------------
// Residual quantization, all 8 levels fused.
// - 4 waves/block, 32 rows/wave (128 rows/block): each codebook pass is
//   amortized over 128 rows -> L2 codebook traffic = 8 levels * 256 blocks
//   * 1MB = 2GB (vs 16GB naive).
// - Codebook tile (16 codes x 256 K, 8KB f16) is double-buffered in LDS:
//   stage tile ct+1 while 16 WMMAs consume tile ct; one barrier per step.
// - f32 residual lives in dynamic LDS (~161KB total; gfx1250 WGP has 320KB).
//   dist = ||cb||^2 - 2*(r.cb); ||r||^2 irrelevant to argmin.
// ---------------------------------------------------------------------------
#define RQ_SMEM_RES   (RQ_ROWS_PER_BLOCK * 256 * 4)          /* 131072 */
#define RQ_SMEM_CBT   (2 * 16 * 256 * 2)                     /*  16384 */
#define RQ_SMEM_RVAL  (RQ_ROWS_PER_BLOCK * 16 * 4)           /*   8192 */
#define RQ_SMEM_RIDX  (RQ_ROWS_PER_BLOCK * 16 * 4)           /*   8192 */
#define RQ_SMEM_SEL   (RQ_ROWS_PER_BLOCK * 4)                /*    512 */
#define RQ_SMEM_SRED  (RQ_WAVES * 32 * 4)                    /*    512 */
#define RQ_SMEM_TOTAL (RQ_SMEM_RES + RQ_SMEM_CBT + RQ_SMEM_RVAL + RQ_SMEM_RIDX + RQ_SMEM_SEL + RQ_SMEM_SRED)

__global__ __launch_bounds__(RQ_WAVES * 32) void rq_kernel(
    const float* __restrict__ z, const _Float16* __restrict__ cb16,
    const float* __restrict__ cbsq, _Float16* __restrict__ zq16,
    float* __restrict__ codes, float* __restrict__ usage,
    float* __restrict__ commit_acc)
{
  extern __shared__ char smem_raw[];
  float*    res  = (float*)smem_raw;                                   // [128][256]
  _Float16* cbt  = (_Float16*)(smem_raw + RQ_SMEM_RES);                // [2][16][256]
  float*    rval = (float*)(smem_raw + RQ_SMEM_RES + RQ_SMEM_CBT);     // [128][16]
  int*      ridx = (int*)((char*)rval + RQ_SMEM_RVAL);                 // [128][16]
  int*      sel  = (int*)((char*)ridx + RQ_SMEM_RIDX);                 // [128]
  float*    sred = (float*)((char*)sel + RQ_SMEM_SEL);                 // [128]

  const int tid  = threadIdx.x;
  const int lane = tid & 31;
  const int wid  = tid >> 5;
  const int gRow = blockIdx.x * RQ_ROWS_PER_BLOCK + wid * RQ_ROWS_PER_WAVE;
  const int wrow0 = wid * RQ_ROWS_PER_WAVE;   // first row of this wave in res[]

  const int NT = KCODES / 16;         // codebook tiles
  const int crow = tid >> 3;          // 0..15 : codebook row within tile
  const int cseg = (tid & 7) * 32;    // 32-half (64B) segment within the row

  // residual := z   (each wave loads its own 32x256 block)
  for (int e = lane; e < (RQ_ROWS_PER_WAVE * 256 / 4); e += 32) {
    int flat = e * 4;
    int r = flat >> 8, c = flat & 255;
    v4f zv = *(const v4f*)(z + (size_t)(gRow + r) * LDIM + c);
    *(v4f*)&res[(wrow0 + r) * 256 + c] = zv;
  }
  __syncthreads();

  float ssq = 0.0f;

  for (int level = 0; level < RQL; ++level) {
    // Build f16 A fragments from f32 residual in LDS: 2 row tiles x 8 K-frags
    v16h af[2][8];
#pragma unroll
    for (int rt = 0; rt < 2; ++rt) {
#pragma unroll
      for (int kc = 0; kc < 8; ++kc) {
        const float* p = &res[(wrow0 + rt * 16 + (lane & 15)) * 256 + kc * 32];
        int ks = (lane & 16) ? 8 : 0;
        v16h f;
#pragma unroll
        for (int i = 0; i < 8; ++i) {
          f[i]     = (_Float16)p[ks + i];
          f[8 + i] = (_Float16)p[ks + 16 + i];
        }
        af[rt][kc] = f;
      }
    }

    float minv[2][8];
    int   mini[2][8];
#pragma unroll
    for (int rt = 0; rt < 2; ++rt)
#pragma unroll
      for (int j = 0; j < 8; ++j) { minv[rt][j] = 3.4e38f; mini[rt][j] = 0; }

    // stage first tile of this level into buffer 0
    {
      const _Float16* src = cb16 + (size_t)crow * 256 + cseg;
      _Float16* d = cbt + crow * 256 + cseg;
#pragma unroll
      for (int i = 0; i < 4; ++i) *(v8h*)(d + i * 8) = *(const v8h*)(src + i * 8);
    }

    for (int ct = 0; ct < NT; ++ct) {
      __syncthreads();  // tile ct staged; readers of buffer (ct+1)&1 are done
      // stage tile ct+1 into the alternate buffer while computing tile ct
      if (ct + 1 < NT) {
        const _Float16* src = cb16 + (size_t)((ct + 1) * 16 + crow) * 256 + cseg;
        _Float16* d = cbt + ((ct + 1) & 1) * (16 * 256) + crow * 256 + cseg;
#pragma unroll
        for (int i = 0; i < 4; ++i) *(v8h*)(d + i * 8) = *(const v8h*)(src + i * 8);
        if (ct + 2 < NT) {
          __builtin_prefetch((const char*)(cb16 + (size_t)((ct + 2) * 16 + crow) * 256 + cseg), 0, 3);
        }
      }

      const _Float16* cur = cbt + (ct & 1) * (16 * 256);
      const int code = ct * 16 + (lane & 15);
      v8f acc0 = zero_v8f();
      v8f acc1 = zero_v8f();
#pragma unroll
      for (int kc = 0; kc < 8; ++kc) {
        const _Float16* p = cur + (lane & 15) * 256 + kc * 32 + ((lane & 16) ? 8 : 0);
        v8h lo = *(const v8h*)p;
        v8h hi = *(const v8h*)(p + 16);
        v16h bfrag;
#pragma unroll
        for (int i = 0; i < 8; ++i) { bfrag[i] = lo[i]; bfrag[8 + i] = hi[i]; }
        acc0 = __builtin_amdgcn_wmma_f32_16x16x32_f16(
            false, af[0][kc], false, bfrag, (short)0, acc0, false, false);
        acc1 = __builtin_amdgcn_wmma_f32_16x16x32_f16(
            false, af[1][kc], false, bfrag, (short)0, acc1, false, false);
      }
      const float cq = cbsq[code];
#pragma unroll
      for (int j = 0; j < 8; ++j) {
        float d0 = cq - 2.0f * acc0[j];
        if (d0 < minv[0][j]) { minv[0][j] = d0; mini[0][j] = code; }
        float d1 = cq - 2.0f * acc1[j];
        if (d1 < minv[1][j]) { minv[1][j] = d1; mini[1][j] = code; }
      }
    }

    // cross-lane argmin via LDS
    const int rbase = (lane & 16) ? 8 : 0;
#pragma unroll
    for (int rt = 0; rt < 2; ++rt) {
#pragma unroll
      for (int j = 0; j < 8; ++j) {
        rval[(wrow0 + rt * 16 + rbase + j) * 16 + (lane & 15)] = minv[rt][j];
        ridx[(wrow0 + rt * 16 + rbase + j) * 16 + (lane & 15)] = mini[rt][j];
      }
    }
    __syncthreads();
    {
      const int r = tid;   // each of the 128 threads reduces one row
      float bv = rval[r * 16 + 0];
      int   bi = ridx[r * 16 + 0];
      for (int c = 1; c < 16; ++c) {
        float v = rval[r * 16 + c];
        int  i2 = ridx[r * 16 + c];
        if (v < bv || (v == bv && i2 < bi)) { bv = v; bi = i2; }
      }
      sel[r] = bi;
      codes[(size_t)(blockIdx.x * RQ_ROWS_PER_BLOCK + r) * RQL + level] = (float)bi;
      atomicAdd(&usage[bi], 1.0f);
    }
    __syncthreads();

    // residual -= cb[idx]; commitment += ||new residual||^2
    for (int e = lane; e < (RQ_ROWS_PER_WAVE * 256 / 4); e += 32) {
      int flat = e * 4;
      int r = flat >> 8, c = flat & 255;
      int code = sel[wrow0 + r];
      v4h q  = *(const v4h*)(cb16 + (size_t)code * LDIM + c);
      v4f rv = *(v4f*)&res[(wrow0 + r) * 256 + c];
#pragma unroll
      for (int i = 0; i < 4; ++i) {
        float nv = rv[i] - (float)q[i];
        rv[i] = nv;
        ssq += nv * nv;
      }
      *(v4f*)&res[(wrow0 + r) * 256 + c] = rv;
    }
    __syncthreads();
  }

  // z_quantized = z - residual_final -> f16 for decoder
  for (int e = lane; e < (RQ_ROWS_PER_WAVE * 256 / 4); e += 32) {
    int flat = e * 4;
    int r = flat >> 8, c = flat & 255;
    v4f zv = *(const v4f*)(z + (size_t)(gRow + r) * LDIM + c);
    v4f rv = *(v4f*)&res[(wrow0 + r) * 256 + c];
    v4h o;
#pragma unroll
    for (int i = 0; i < 4; ++i) o[i] = (_Float16)(zv[i] - rv[i]);
    *(v4h*)(zq16 + (size_t)(gRow + r) * LDIM + c) = o;
  }

  sred[tid] = ssq;
  __syncthreads();
  if (tid == 0) {
    float s = 0.0f;
    for (int i = 0; i < RQ_WAVES * 32; ++i) s += sred[i];
    atomicAdd(commit_acc, s);
  }
}

// ---------------------------------------------------------------------------
// Prep / finalize kernels
// ---------------------------------------------------------------------------
__global__ void transpose_w_kernel(const float* __restrict__ in, _Float16* __restrict__ out,
                                   int Kd, int Nd) {  // in [Kd][Nd] -> out [Nd][Kd]
  int i = blockIdx.x * 256 + threadIdx.x;
  if (i < Kd * Nd) {
    int n = i / Kd, k = i - n * Kd;
    out[i] = (_Float16)in[(size_t)k * Nd + n];
  }
}

__global__ void cb_convert_kernel(const float* __restrict__ cb, _Float16* __restrict__ cb16) {
  int i = blockIdx.x * 256 + threadIdx.x;
  if (i < KCODES * LDIM) cb16[i] = (_Float16)cb[i];
}

__global__ void cbsq_kernel(const float* __restrict__ cb, float* __restrict__ cbsq) {
  int j = blockIdx.x * 256 + threadIdx.x;
  if (j < KCODES) {
    float s = 0.0f;
    const float* p = cb + (size_t)j * LDIM;
    for (int k = 0; k < LDIM; ++k) { float v = p[k]; s += v * v; }
    cbsq[j] = s;
  }
}

__global__ void init_kernel(float* accum, float* usage) {
  int i = blockIdx.x * 256 + threadIdx.x;
  if (i < 2) accum[i] = 0.0f;
  if (i < KCODES) usage[i] = 0.0f;
}

__global__ void finalize_kernel(const float* accum, float* out) {
  float recon  = accum[0] / ((float)BSZ * (float)DDIM);
  float commit = accum[1] / ((float)BSZ * (float)LDIM);
  out[0] = recon + 0.25f * commit;
  out[1] = recon;
  out[2] = commit;
}

// ---------------------------------------------------------------------------
extern "C" void kernel_launch(void* const* d_in, const int* in_sizes, int n_in,
                              void* d_out, int out_size, void* d_ws, size_t ws_size,
                              hipStream_t stream)
{
  const float* x      = (const float*)d_in[0];
  const float* enc_w1 = (const float*)d_in[1];
  const float* enc_b1 = (const float*)d_in[2];
  const float* enc_w2 = (const float*)d_in[3];
  const float* enc_b2 = (const float*)d_in[4];
  const float* cb     = (const float*)d_in[5];
  const float* dec_w1 = (const float*)d_in[6];
  const float* dec_b1 = (const float*)d_in[7];
  const float* dec_w2 = (const float*)d_in[8];
  const float* dec_b2 = (const float*)d_in[9];
  float* out = (float*)d_out;

  char* w = (char*)d_ws;
  auto take = [&](size_t bytes) {
    void* p = (void*)w;
    w += (bytes + 255) & ~(size_t)255;
    return p;
  };
  _Float16* ew1t = (_Float16*)take((size_t)HDIM * DDIM * 2);
  _Float16* ew2t = (_Float16*)take((size_t)LDIM * HDIM * 2);
  _Float16* dw1t = (_Float16*)take((size_t)HDIM * LDIM * 2);
  _Float16* dw2t = (_Float16*)take((size_t)DDIM * HDIM * 2);
  _Float16* cb16 = (_Float16*)take((size_t)KCODES * LDIM * 2);
  float*    cbsq = (float*)take((size_t)KCODES * 4);
  _Float16* a1   = (_Float16*)take((size_t)BSZ * HDIM * 2);
  float*    zbuf = (float*)take((size_t)BSZ * LDIM * 4);
  _Float16* zq16 = (_Float16*)take((size_t)BSZ * LDIM * 2);
  float*    accum = (float*)take(256);

  float* codes = out + 3;
  float* usage = out + 3 + (size_t)BSZ * RQL;

  init_kernel<<<(KCODES + 255) / 256, 256, 0, stream>>>(accum, usage);
  transpose_w_kernel<<<((DDIM * HDIM) + 255) / 256, 256, 0, stream>>>(enc_w1, ew1t, DDIM, HDIM);
  transpose_w_kernel<<<((HDIM * LDIM) + 255) / 256, 256, 0, stream>>>(enc_w2, ew2t, HDIM, LDIM);
  transpose_w_kernel<<<((LDIM * HDIM) + 255) / 256, 256, 0, stream>>>(dec_w1, dw1t, LDIM, HDIM);
  transpose_w_kernel<<<((HDIM * DDIM) + 255) / 256, 256, 0, stream>>>(dec_w2, dw2t, HDIM, DDIM);
  cb_convert_kernel<<<((KCODES * LDIM) + 255) / 256, 256, 0, stream>>>(cb, cb16);
  cbsq_kernel<<<(KCODES + 255) / 256, 256, 0, stream>>>(cb, cbsq);

  { dim3 g(HDIM / 128, BSZ / 128);
    gemm_kernel<false, true, 0><<<g, 256, 0, stream>>>(x, ew1t, enc_b1, a1, nullptr, nullptr, BSZ, HDIM, DDIM); }
  { dim3 g(LDIM / 128, BSZ / 128);
    gemm_kernel<true, false, 1><<<g, 256, 0, stream>>>(a1, ew2t, enc_b2, zbuf, nullptr, nullptr, BSZ, LDIM, HDIM); }

  rq_kernel<<<BSZ / RQ_ROWS_PER_BLOCK, RQ_WAVES * 32, RQ_SMEM_TOTAL, stream>>>(
      zbuf, cb16, cbsq, zq16, codes, usage, accum + 1);

  { dim3 g(HDIM / 128, BSZ / 128);
    gemm_kernel<true, true, 0><<<g, 256, 0, stream>>>(zq16, dw1t, dec_b1, a1, nullptr, nullptr, BSZ, HDIM, LDIM); }
  { dim3 g(DDIM / 128, BSZ / 128);
    gemm_kernel<true, false, 2><<<g, 256, 0, stream>>>(a1, dw2t, dec_b2, nullptr, x, accum + 0, BSZ, DDIM, HDIM); }

  finalize_kernel<<<1, 1, 0, stream>>>(accum, out);
}
